// SparseLayer_73650099192114
// MI455X (gfx1250) — matrix operand order repaired and checked by reading, not compiled, not run on hardware
//
#include <hip/hip_runtime.h>

#define IN_DIM  27687
#define OUT_DIM 1387
#define NNZ     71000
#define BATCH   4096
#define NT      ((OUT_DIM + 15) / 16)   /* 87 column tiles */

#define KSTAGE  128                     /* nnz staged per barrier round */
#define BT_PITCH 136                    /* ushorts per BT row: 272B = 17*16B, bank-conflict-free */

typedef __attribute__((ext_vector_type(16))) __bf16 v16bf;
typedef __attribute__((ext_vector_type(8)))  float  v8f;

__device__ __forceinline__ unsigned short f32_to_bf16_bits(float f) {
    unsigned int u = __float_as_uint(f);
    unsigned int r = (u + 0x7FFFu + ((u >> 16) & 1u)) >> 16;
    return (unsigned short)r;
}

// ---- Phase 1: histogram of nnz per 16-column output tile -------------------
__global__ void bin_count_kernel(const int* __restrict__ idx,
                                 int* __restrict__ counts) {
    int k = blockIdx.x * blockDim.x + threadIdx.x;
    if (k < NNZ) {
        int j = idx[2 * k + 1];
        atomicAdd(&counts[j >> 4], 1);
    }
}

// ---- Phase 2: exclusive prefix sum over 87 bins (trivial, one lane) --------
__global__ void prefix_kernel(const int* __restrict__ counts,
                              int* __restrict__ offsets,
                              int* __restrict__ cursor) {
    if (threadIdx.x == 0) {
        int acc = 0;
        for (int t = 0; t < NT; ++t) {
            offsets[t] = acc;
            cursor[t]  = acc;
            acc += counts[t];
        }
        offsets[NT] = acc;
    }
}

// ---- Phase 3: scatter nnz into tile-sorted order (packed gene|col) ---------
__global__ void bin_scatter_kernel(const int* __restrict__ idx,
                                   const float* __restrict__ w,
                                   int* __restrict__ cursor,
                                   int* __restrict__ sp,     // ii | (jl<<16)
                                   float* __restrict__ sw) {
    int k = blockIdx.x * blockDim.x + threadIdx.x;
    if (k < NNZ) {
        int i = idx[2 * k + 0];          // < 27687, fits in 15 bits
        int j = idx[2 * k + 1];
        int pos = atomicAdd(&cursor[j >> 4], 1);
        sp[pos] = i | ((j & 15) << 16);
        sw[pos] = w[k];
    }
}

// ---- Phase 4: binned SpMM via v_wmma_f32_16x16x32_bf16 ---------------------
// Grid: (BATCH/128, NT). Block: 256 threads = 8 waves; each wave owns 16 batch
// rows. Per barrier round: stage 128 nnz + one-hot B^T in LDS, then each wave
// issues 4 WMMAs (K=128) fed by 64 global bf16 gathers.
__global__ __launch_bounds__(256)
void spmm_wmma_kernel(const float* __restrict__ x,
                      const int* __restrict__ sp,
                      const float* __restrict__ sw,
                      const int* __restrict__ offsets,
                      float* __restrict__ out) {
    __shared__ int                           ls_i[KSTAGE];         // gene indices
    __shared__ __align__(16) unsigned short  ls_bt[16][BT_PITCH];  // B^T: [n][k]

    const int tid  = threadIdx.x;
    const int lane = tid & 31;
    const int wave = tid >> 5;
    const int tile = blockIdx.y;
    const int m0   = blockIdx.x * 128 + wave * 16;

    const int kbeg = offsets[tile];
    const int kend = offsets[tile + 1];

    const int nloc = lane & 15;     // A: M index within wave tile; B: N index
    const int hi   = lane >> 4;
    const size_t xrow = (size_t)(m0 + nloc) * IN_DIM;

    v8f acc = {};

    for (int base = kbeg; base < kend; base += KSTAGE) {
        __syncthreads();   // previous round's LDS reads complete

        // branchless cooperative zero of B^T: 16*136 ushorts = 272 x 16B
        {
            uint4* btv = (uint4*)&ls_bt[0][0];
            uint4  z4  = {0u, 0u, 0u, 0u};
            btv[tid] = z4;
            if (tid < 16 * BT_PITCH / 8 - 256) btv[256 + tid] = z4;
        }

        // threads 0..127 each load one packed (gene|col, w) pair
        int  ii = 0, jl = 0;
        float wv = 0.0f;
        const bool have = (tid < KSTAGE) && (base + tid < kend);
        if (have) {
            int k  = base + tid;
            int pk = sp[k];
            ii = pk & 0xFFFF;
            jl = pk >> 16;
            wv = sw[k];
            __builtin_prefetch((const void*)(sp + k + KSTAGE), 0, 1);
        }
        if (tid < KSTAGE) ls_i[tid] = ii;

        __syncthreads();   // zeroing complete
        if (have) ls_bt[jl][tid] = f32_to_bf16_bits(wv);   // one-hot row k
        __syncthreads();   // B^T ready

        const int nsub = min(4, (kend - base + 31) >> 5);
        for (int sub = 0; sub < nsub; ++sub) {
            union { v16bf v; __bf16 h[16]; unsigned short u[16]; } afrag, bfrag;
            const int kb = sub * 32;

            // B fragment (32x16): lanes 0-15 -> K=kb..kb+15, lanes 16-31 -> +16
            #pragma unroll
            for (int e = 0; e < 16; ++e)
                bfrag.u[e] = ls_bt[nloc][kb + hi * 16 + e];

            // A fragment (16x32): gather x[m, idx0[k]], native bf16 convert
            #pragma unroll
            for (int e = 0; e < 16; ++e) {
                int kc = kb + hi * 16 + e;
                afrag.h[e] = (__bf16)x[xrow + ls_i[kc]];
            }

            acc = __builtin_amdgcn_wmma_f32_16x16x32_bf16(
                false, afrag.v, false, bfrag.v, (short)0, acc, false, false);
        }
    }

    // C/D layout: VGPR r -> (M = r + 8*hi, N = lane&15)
    const int n = tile * 16 + nloc;
    if (n < OUT_DIM) {
        const int mbase = m0 + hi * 8;
        #pragma unroll
        for (int r = 0; r < 8; ++r)
            out[(size_t)(mbase + r) * OUT_DIM + n] = acc[r];
    }
}

extern "C" void kernel_launch(void* const* d_in, const int* in_sizes, int n_in,
                              void* d_out, int out_size, void* d_ws, size_t ws_size,
                              hipStream_t stream) {
    const float* x   = (const float*)d_in[0];
    const int*   idx = (const int*)d_in[1];   // [NNZ,2] int32 (gene, pathway)
    const float* w   = (const float*)d_in[2];
    float*       out = (float*)d_out;

    // Workspace layout (~0.57 MB): meta (3x128 ints) + binned packed pairs
    int*   meta    = (int*)d_ws;
    int*   counts  = meta;          // NT used
    int*   offsets = meta + 128;    // NT+1 used
    int*   cursor  = meta + 256;    // NT used
    int*   sp      = meta + 384;    // NNZ packed (gene | col<<16)
    float* sw      = (float*)(sp + NNZ);

    hipMemsetAsync(meta, 0, 384 * sizeof(int), stream);

    bin_count_kernel<<<(NNZ + 255) / 256, 256, 0, stream>>>(idx, counts);
    prefix_kernel<<<1, 32, 0, stream>>>(counts, offsets, cursor);
    bin_scatter_kernel<<<(NNZ + 255) / 256, 256, 0, stream>>>(idx, w, cursor, sp, sw);

    spmm_wmma_kernel<<<dim3(BATCH / 128, NT), 256, 0, stream>>>(
        x, sp, sw, offsets, out);
}
